// RNN_48017734369491
// MI455X (gfx1250) — compile-verified, matrix operand order checked by reading, not run on hardware
//
#include <hip/hip_runtime.h>

// ---------------------------------------------------------------------------
// LSTM classifier on MI455X (gfx1250, wave32, WMMA, LDS-staged weights).
//
//   1) build_wT   : fuse Wx[256,2048] + Wh[512,2048] -> WcT bf16 [2048][768]
//   2) zero_state : h0 (bf16) = 0, c (f32) = 0
//   3) 256 x lstm_step : one fused kernel per time step.
//        block  = one unit tile (16 units x 4 gates), 8 waves = 8 batch tiles
//        weight slice staged through LDS in three 256-K stages that align with
//        the semantic K split (stage 0 = embedding part, 1-2 = recurrent), so
//        the A-source choice is compile-time and the loop is branch-free.
//        Fully unrolled: 24 K-iters x 4 v_wmma_f32_16x16x32_bf16 per wave,
//        A fragments preloaded per stage, fused LSTM cell epilogue.
//   4) write_ones : softmax over a size-1 axis == 1.0 exactly.
// ---------------------------------------------------------------------------

#define VOCAB 32000
#define EMB   256
#define UNITS 512
#define SEQ   256
#define BATCH 128
#define KDIM  (EMB + UNITS)    // 768 fused reduction dim
#define NGATE (4 * UNITS)      // 2048 gate columns

#define KSTAGE    256          // K elements staged per stage (3 stages)
#define LDS_PITCH (KSTAGE + 8) // padded row pitch in bf16: rows hit distinct
                               // bank groups (132 dwords ≡ 4 mod 64 per row)

typedef __attribute__((ext_vector_type(16))) __bf16 v16bf;
typedef __attribute__((ext_vector_type(8)))  float  v8f;

__device__ __forceinline__ float sigmoidf(float x) {
    return 1.0f / (1.0f + __expf(-x));
}

// Build transposed fused bf16 weight: WcT[col][k], col in [0,2048), k in [0,768).
__global__ void build_wT(const float* __restrict__ Wx,
                         const float* __restrict__ Wh,
                         __bf16* __restrict__ WcT) {
    int idx = blockIdx.x * blockDim.x + threadIdx.x;   // over 2048*768
    if (idx >= NGATE * KDIM) return;
    int c = idx / KDIM;
    int k = idx - c * KDIM;
    float v = (k < EMB) ? Wx[k * NGATE + c] : Wh[(k - EMB) * NGATE + c];
    WcT[idx] = (__bf16)v;
}

__global__ void zero_state(__bf16* __restrict__ h, float* __restrict__ c) {
    int i = blockIdx.x * blockDim.x + threadIdx.x;
    if (i < BATCH * UNITS) { h[i] = (__bf16)0.0f; c[i] = 0.0f; }
}

// One LSTM time step. Grid: 32 blocks (= unit tiles) x 256 threads (8 waves =
// 8 batch tiles). B fragments served from LDS; A gathered per wave.
__global__ __launch_bounds__(256) void lstm_step(
    const int*    __restrict__ sentence,   // [BATCH][SEQ] int32
    const float*  __restrict__ emb,        // [VOCAB][EMB] f32
    const __bf16* __restrict__ WcT,        // [NGATE][KDIM] bf16 (transposed)
    const float*  __restrict__ bias,       // [NGATE] f32
    const __bf16* __restrict__ h_in,       // [BATCH][UNITS] bf16
    __bf16*       __restrict__ h_out,      // [BATCH][UNITS] bf16
    float*        __restrict__ c_state,    // [BATCH][UNITS] f32 (in place)
    int t) {
    // 64 rows (4 gates x 16 cols) x padded pitch, one 256-K stage resident.
    __shared__ __bf16 sB[64 * LDS_PITCH];  // 33,792 bytes

    const int utile = blockIdx.x;          // 0..31 unit tile
    const int mtile = threadIdx.x >> 5;    // 0..7  batch tile (wave id)
    const int lane  = threadIdx.x & 31;

    // ISA 7.12.2 16-bit layouts: lanes 0-15 hold K {0..7, 16..23},
    // lanes 16-31 hold K {8..15, 24..31} of the same 16 rows/cols.
    const int row  = mtile * 16 + (lane & 15);        // batch row for A
    const int koff = (lane & 16) ? 8 : 0;
    const int un   = utile * 16 + (lane & 15);        // unit column for C/D

    v8f acc[4];
    #pragma unroll
    for (int g = 0; g < 4; ++g) acc[g] = v8f{};

    const int token = sentence[row * SEQ + t];
    const float*  embrow = emb  + (size_t)token * EMB;
    const __bf16* hrow   = h_in + (size_t)row * UNITS;

    #pragma unroll
    for (int s = 0; s < 3; ++s) {                 // s is compile-time constant
        const int k0 = s * KSTAGE;
        if (s) __syncthreads();                   // previous stage consumed
        // ---- cooperative stage: 64 rows x 32 x 16B chunks -> LDS ----
        #pragma unroll
        for (int i = 0; i < (64 * KSTAGE / 8) / 256; ++i) {
            const int cch = i * 256 + threadIdx.x;
            const int r   = cch >> 5;             // 0..63 = g*16 + col
            const int kc  = (cch & 31) * 8;       // bf16 offset in stage
            const int g   = r >> 4;
            const int col = r & 15;
            const __bf16* src =
                WcT + (size_t)(g * UNITS + utile * 16 + col) * KDIM + k0 + kc;
            *(uint4*)&sB[r * LDS_PITCH + kc] = *(const uint4*)src;
        }
        __syncthreads();

        // ---- preload the stage's 8 A fragments (uniform source: s==0 is
        //      the embedding part, s>=1 the recurrent part) ----
        v16bf a[8];
        #pragma unroll
        for (int kk = 0; kk < 8; ++kk) {
            const int kb = k0 + kk * 32;
            if (s == 0) {
                const float* p = embrow + kb + koff;
                #pragma unroll
                for (int e = 0; e < 8; ++e) {
                    a[kk][e]     = (__bf16)p[e];
                    a[kk][8 + e] = (__bf16)p[16 + e];
                }
            } else {
                const __bf16* p = hrow + (kb - EMB) + koff;
                #pragma unroll
                for (int e = 0; e < 8; ++e) {
                    a[kk][e]     = p[e];
                    a[kk][8 + e] = p[16 + e];
                }
            }
        }

        // ---- 8 K-iterations against the staged slice, fully unrolled ----
        #pragma unroll
        for (int kk = 0; kk < 8; ++kk) {
            const int kl = kk * 32;               // K offset within stage
            v16bf bfr[4];
            #pragma unroll
            for (int g = 0; g < 4; ++g) {
                const __bf16* p =
                    &sB[(g * 16 + (lane & 15)) * LDS_PITCH + kl + koff];
                #pragma unroll
                for (int e = 0; e < 8; ++e) {
                    bfr[g][e]     = p[e];
                    bfr[g][8 + e] = p[16 + e];
                }
            }
            #pragma unroll
            for (int g = 0; g < 4; ++g) {
                acc[g] = __builtin_amdgcn_wmma_f32_16x16x32_bf16(
                    false, a[kk], false, bfr[g], (short)0, acc[g], false, false);
            }
        }
    }

    // ---- fused LSTM cell epilogue ----
    // C/D layout: VGPR v, lanes 0-15 -> M = v, lanes 16-31 -> M = v + 8.
    const float bi  = bias[0 * UNITS + un];
    const float bf_ = bias[1 * UNITS + un];
    const float bg  = bias[2 * UNITS + un];
    const float bo  = bias[3 * UNITS + un];
    const int mbase = mtile * 16 + ((lane & 16) ? 8 : 0);
    #pragma unroll
    for (int v = 0; v < 8; ++v) {
        const int m   = mbase + v;
        const int idx = m * UNITS + un;
        const float i_ = sigmoidf(acc[0][v] + bi);
        const float f_ = sigmoidf(acc[1][v] + bf_);
        const float g_ = tanhf   (acc[2][v] + bg);
        const float o_ = sigmoidf(acc[3][v] + bo);
        const float cn = f_ * c_state[idx] + i_ * g_;
        c_state[idx] = cn;
        h_out[idx]   = (__bf16)(o_ * tanhf(cn));
    }
}

// softmax over a length-1 axis is identically 1.0 (exact, not approximate).
__global__ void write_ones(float* __restrict__ out, int n) {
    int i = blockIdx.x * blockDim.x + threadIdx.x;
    if (i < n) out[i] = 1.0f;
}

extern "C" void kernel_launch(void* const* d_in, const int* in_sizes, int n_in,
                              void* d_out, int out_size, void* d_ws, size_t ws_size,
                              hipStream_t stream) {
    const int*   sentence = (const int*)  d_in[0];
    const float* emb      = (const float*)d_in[1];
    const float* Wx       = (const float*)d_in[2];
    const float* Wh       = (const float*)d_in[3];
    const float* b        = (const float*)d_in[4];
    // d_in[5..8] (W1,b1,W2,b2) are mathematically dead: softmax over the
    // size-1 logits axis yields exactly 1.0 for every batch row.

    char* ws = (char*)d_ws;
    __bf16* WcT = (__bf16*)ws;                                  // 3,145,728 B
    __bf16* h0  = (__bf16*)(ws + (size_t)NGATE * KDIM * 2);     //   131,072 B
    __bf16* h1  = h0 + BATCH * UNITS;                           //   131,072 B
    float*  c   = (float*)(h1 + BATCH * UNITS);                 //   262,144 B

    build_wT<<<(NGATE * KDIM + 255) / 256, 256, 0, stream>>>(Wx, Wh, WcT);
    zero_state<<<(BATCH * UNITS + 255) / 256, 256, 0, stream>>>(h0, c);

    for (int t = 0; t < SEQ; ++t) {
        const __bf16* hin  = (t & 1) ? h1 : h0;
        __bf16*       hout = (t & 1) ? h0 : h1;
        lstm_step<<<32, 256, 0, stream>>>(sentence, emb, WcT, b, hin, hout, c, t);
    }

    write_ones<<<1, 128, 0, stream>>>((float*)d_out, out_size);
}
